// GravityCausalWrapper_89464168775877
// MI455X (gfx1250) — compile-verified, matrix operand order checked by reading, not run on hardware
//
#include <hip/hip_runtime.h>
#include <hip/hip_bf16.h>

#define N_NODES 100000
#define N_EDGES 1600000
#define EMB 64
#define PHI_H 32
#define PRED_H 64

typedef _Float16 v16h __attribute__((ext_vector_type(16)));
typedef _Float16 v8h  __attribute__((ext_vector_type(8)));
typedef float    v8f  __attribute__((ext_vector_type(8)));

union V16U { v16h v; v8h h[2]; };

// ---------------------------------------------------------------------------
// Kernel 1: zero agg/deg accumulator and pack Wp1 (64x32 f32) into the
// per-lane f16 B-operand layout for v_wmma_f32_16x16x32_f16.
// wp1p index = ((t*2 + c)*32 + lane)*16 + j   (t = N-tile, c = K-chunk)
//   lane<16 : N = 16t + lane,    K = 32c + { j<8 ? j : 16+(j-8) }
//   lane>=16: N = 16t + lane-16, K = 32c + { j<8 ? 8+j : 24+(j-8) }
// ---------------------------------------------------------------------------
__global__ void k_init(float* __restrict__ aggdeg,
                       const float* __restrict__ Wp1,
                       _Float16* __restrict__ wp1p) {
    int idx = blockIdx.x * blockDim.x + threadIdx.x;
    if (idx < N_NODES * 4) aggdeg[idx] = 0.0f;
    if (idx < 2048) {
        int j = idx & 15;
        int l = (idx >> 4) & 31;
        int c = (idx >> 9) & 1;
        int t = (idx >> 10) & 1;
        int hs = (l >> 4) * 8;
        int kk = (j < 8) ? (hs + j) : (16 + hs + (j - 8));
        int k  = 32 * c + kk;
        int n  = 16 * t + (l & 15);
        wp1p[idx] = (_Float16)Wp1[k * PHI_H + n];
    }
}

// ---------------------------------------------------------------------------
// Kernel 2: scatter-add x[src] and a count into aggdeg[dst] (float4 per node)
// ---------------------------------------------------------------------------
__global__ void k_scatter(const float* __restrict__ x,
                          const long long* __restrict__ src,
                          const long long* __restrict__ dst,
                          float* __restrict__ aggdeg) {
    int e = blockIdx.x * blockDim.x + threadIdx.x;
    if (e >= N_EDGES) return;
    long long s = src[e];
    long long d = dst[e];
    const float* xs = x + s * 3;
    float* a = aggdeg + d * 4;
    atomicAdd(a + 0, xs[0]);
    atomicAdd(a + 1, xs[1]);
    atomicAdd(a + 2, xs[2]);
    atomicAdd(a + 3, 1.0f);
}

// ---------------------------------------------------------------------------
// Kernel 3: node embedding  x_emb = relu(x@W_self + (agg/deg)@W_nbr + b)
// one thread per (node, j); stores f16 full embedding + float4 of dims 0..2
// ---------------------------------------------------------------------------
__global__ void k_embed(const float* __restrict__ x,
                        const float* __restrict__ aggdeg,
                        const float* __restrict__ W_self,
                        const float* __restrict__ W_nbr,
                        const float* __restrict__ b_gnn,
                        _Float16* __restrict__ emb_h,
                        float* __restrict__ emb4) {
    int id = blockIdx.x * blockDim.x + threadIdx.x;
    if (id >= N_NODES * EMB) return;
    int n = id >> 6;
    int j = id & 63;
    const float* ad = aggdeg + n * 4;
    float invd = 1.0f / fmaxf(ad[3], 1.0f);
    float a0 = ad[0] * invd, a1 = ad[1] * invd, a2 = ad[2] * invd;
    const float* xs = x + n * 3;
    float v = b_gnn[j];
    v = fmaf(xs[0], W_self[0 * EMB + j], v);
    v = fmaf(xs[1], W_self[1 * EMB + j], v);
    v = fmaf(xs[2], W_self[2 * EMB + j], v);
    v = fmaf(a0, W_nbr[0 * EMB + j], v);
    v = fmaf(a1, W_nbr[1 * EMB + j], v);
    v = fmaf(a2, W_nbr[2 * EMB + j], v);
    v = fmaxf(v, 0.0f);
    emb_h[id] = (_Float16)v;
    if (j < 4) emb4[n * 4 + j] = (j < 3) ? v : 0.0f;
}

// ---------------------------------------------------------------------------
// Kernel 4: per-edge fused kernel. Wave32 owns 32 edges.
//   phi hidden layer (64->32) via 8x v_wmma_f32_16x16x32_f16 per wave,
//   spill D tiles through LDS, then per-lane scalar epilogue:
//   phi dot, causal features, 6->64->1 MLP, sigmoid.
// ---------------------------------------------------------------------------
__global__ void __launch_bounds__(256)
k_edges(const long long* __restrict__ src,
        const long long* __restrict__ dst,
        const _Float16* __restrict__ emb_h,
        const float* __restrict__ emb4,
        const _Float16* __restrict__ wp1p,
        const float* __restrict__ bp1,
        const float* __restrict__ Wp2,
        const float* __restrict__ bp2,
        const float* __restrict__ We1,
        const float* __restrict__ be1,
        const float* __restrict__ We2,
        const float* __restrict__ be2,
        float* __restrict__ out) {
    __shared__ float hid[8][32][PHI_H];   // per-wave spill of D tiles (32 KB)

    const int lane = threadIdx.x & 31;
    const int wave = threadIdx.x >> 5;
    const int base32 = blockIdx.x * 256 + wave * 32;   // first edge of this wave
    const int hs = (lane >> 4) * 8;                    // lane-half K offset
    const int col = lane & 15;

    // Loop-invariant B operands: 4 tiles (t in {0,1} x c in {0,1})
    V16U b00, b01, b10, b11;
    {
        const v16h* bp = (const v16h*)wp1p;
        b00.v = bp[(0 * 32) + lane];       // t=0,c=0
        b01.v = bp[(1 * 32) + lane];       // t=0,c=1
        b10.v = bp[(2 * 32) + lane];       // t=1,c=0
        b11.v = bp[(3 * 32) + lane];       // t=1,c=1
    }

    // Two 16-edge A tiles -> 4 accumulators (edge-tile T x N-tile t)
    v8f acc00 = {}, acc01 = {}, acc10 = {}, acc11 = {};

#pragma unroll
    for (int T = 0; T < 2; ++T) {
        int row = col;                          // M within tile for this lane
        int e = base32 + T * 16 + row;
        e = (e < N_EDGES) ? e : (N_EDGES - 1);  // clamp: keep EXEC all-ones
        long long s = src[e];
        const _Float16* ep = emb_h + (long long)s * EMB;

        V16U a0, a1;
        a0.h[0] = *(const v8h*)(ep + 0 + hs);        // K 0..7 / 8..15
        a0.h[1] = *(const v8h*)(ep + 16 + hs);        // K 16..23 / 24..31
        a1.h[0] = *(const v8h*)(ep + 32 + hs);        // K 32..39 / 40..47
        a1.h[1] = *(const v8h*)(ep + 48 + hs);        // K 48..55 / 56..63

        if (T == 0) {
            acc00 = __builtin_amdgcn_wmma_f32_16x16x32_f16(false, a0.v, false, b00.v, (short)0, acc00, false, false);
            acc00 = __builtin_amdgcn_wmma_f32_16x16x32_f16(false, a1.v, false, b01.v, (short)0, acc00, false, false);
            acc01 = __builtin_amdgcn_wmma_f32_16x16x32_f16(false, a0.v, false, b10.v, (short)0, acc01, false, false);
            acc01 = __builtin_amdgcn_wmma_f32_16x16x32_f16(false, a1.v, false, b11.v, (short)0, acc01, false, false);
        } else {
            acc10 = __builtin_amdgcn_wmma_f32_16x16x32_f16(false, a0.v, false, b00.v, (short)0, acc10, false, false);
            acc10 = __builtin_amdgcn_wmma_f32_16x16x32_f16(false, a1.v, false, b01.v, (short)0, acc10, false, false);
            acc11 = __builtin_amdgcn_wmma_f32_16x16x32_f16(false, a0.v, false, b10.v, (short)0, acc11, false, false);
            acc11 = __builtin_amdgcn_wmma_f32_16x16x32_f16(false, a1.v, false, b11.v, (short)0, acc11, false, false);
        }
    }

    // Spill D tiles to LDS: lane<16 holds rows 0..7 at N=lane; lane>=16 rows 8..15
    {
        const int rbase = (lane >> 4) * 8;
#pragma unroll
        for (int v = 0; v < 8; ++v) {
            hid[wave][0 * 16 + rbase + v][0 * 16 + col] = acc00[v];
            hid[wave][0 * 16 + rbase + v][1 * 16 + col] = acc01[v];
            hid[wave][1 * 16 + rbase + v][0 * 16 + col] = acc10[v];
            hid[wave][1 * 16 + rbase + v][1 * 16 + col] = acc11[v];
        }
    }
    __builtin_amdgcn_s_barrier();   // conservative; each wave reads its own region

    // ---- per-lane scalar epilogue: this lane owns edge (base32 + lane) ----
    int e = base32 + lane;
    int ec = (e < N_EDGES) ? e : (N_EDGES - 1);

    float phi = bp2[0];
#pragma unroll
    for (int n = 0; n < PHI_H; ++n) {
        float h = hid[wave][lane][n] + bp1[n];
        h = fmaxf(h, 0.0f);
        phi = fmaf(h, Wp2[n], phi);
    }

    long long s = src[ec];
    long long d = dst[ec];
    const float4 ei = *(const float4*)(emb4 + s * 4);
    const float4 ej = *(const float4*)(emb4 + d * 4);
    float dt = ej.x - ei.x;
    float dx = ej.y - ei.y;
    float dy = ej.z - ei.z;
    float dx2 = fmaf(dx, dx, dy * dy);
    float dt2 = dt * dt;
    float ds2 = fmaf(-phi, dt2, dx2);
    float sd  = sqrtf(dx2);
    float itl = (ds2 < 0.0f) ? 1.0f : 0.0f;

    // feats = [dx, dy, dt, ds2, sd, itl] ; logits = relu(feats@We1+be1)@We2+be2
    float logit = be2[0];
#pragma unroll 8
    for (int u = 0; u < PRED_H; ++u) {
        float h = be1[u];
        h = fmaf(dx,  We1[0 * PRED_H + u], h);
        h = fmaf(dy,  We1[1 * PRED_H + u], h);
        h = fmaf(dt,  We1[2 * PRED_H + u], h);
        h = fmaf(ds2, We1[3 * PRED_H + u], h);
        h = fmaf(sd,  We1[4 * PRED_H + u], h);
        h = fmaf(itl, We1[5 * PRED_H + u], h);
        h = fmaxf(h, 0.0f);
        logit = fmaf(h, We2[u], logit);
    }

    if (e < N_EDGES) out[e] = 1.0f / (1.0f + __expf(-logit));
}

// ---------------------------------------------------------------------------
extern "C" void kernel_launch(void* const* d_in, const int* in_sizes, int n_in,
                              void* d_out, int out_size, void* d_ws, size_t ws_size,
                              hipStream_t stream) {
    const float*     x      = (const float*)d_in[0];
    const long long* eidx   = (const long long*)d_in[1];
    const float*     W_self = (const float*)d_in[2];
    const float*     W_nbr  = (const float*)d_in[3];
    const float*     b_gnn  = (const float*)d_in[4];
    const float*     Wp1    = (const float*)d_in[5];
    const float*     bp1    = (const float*)d_in[6];
    const float*     Wp2    = (const float*)d_in[7];
    const float*     bp2    = (const float*)d_in[8];
    const float*     We1    = (const float*)d_in[9];
    const float*     be1    = (const float*)d_in[10];
    const float*     We2    = (const float*)d_in[11];
    const float*     be2    = (const float*)d_in[12];
    float* out = (float*)d_out;

    const long long* src = eidx;
    const long long* dst = eidx + N_EDGES;

    // workspace layout (256B aligned)
    char* ws = (char*)d_ws;
    size_t o = 0;
    float* aggdeg = (float*)(ws + o);           o += (size_t)N_NODES * 4 * sizeof(float);
    o = (o + 255) & ~(size_t)255;
    _Float16* emb_h = (_Float16*)(ws + o);      o += (size_t)N_NODES * EMB * sizeof(_Float16);
    o = (o + 255) & ~(size_t)255;
    float* emb4 = (float*)(ws + o);             o += (size_t)N_NODES * 4 * sizeof(float);
    o = (o + 255) & ~(size_t)255;
    _Float16* wp1p = (_Float16*)(ws + o);       o += 2048 * sizeof(_Float16);
    (void)o; (void)ws_size; (void)n_in; (void)in_sizes; (void)out_size;

    k_init<<<(N_NODES * 4 + 255) / 256, 256, 0, stream>>>(aggdeg, Wp1, wp1p);
    k_scatter<<<(N_EDGES + 255) / 256, 256, 0, stream>>>(x, src, dst, aggdeg);
    k_embed<<<(N_NODES * EMB + 255) / 256, 256, 0, stream>>>(
        x, aggdeg, W_self, W_nbr, b_gnn, emb_h, emb4);
    k_edges<<<N_EDGES / 256, 256, 0, stream>>>(
        src, dst, emb_h, emb4, wp1p, bp1, Wp2, bp2, We1, be1, We2, be2, out);
}
